// RNA3DTransformerGNN_47863115546711
// MI455X (gfx1250) — matrix operand order repaired and checked by reading, not compiled, not run on hardware
//
#include <hip/hip_runtime.h>
#include <hip/hip_bf16.h>

// ---------------------------------------------------------------------------
// RNA3DTransformerGNN for MI455X (gfx1250, wave32, WMMA).
// GEMMs: 64x64 block tile, 8 waves, 2x v_wmma_f32_16x16x32_f16 per wave per
// K-step (shared A fragment), f32->f16 LDS staging, f32 accumulate, prefetch
// of next K slice. Attention: fused flash-style, 32-key step = 2 score WMMAs
// + full-K P.V WMMAs (no zero padding). GAT edge softmax: f32 global atomics.
// ---------------------------------------------------------------------------

typedef __attribute__((ext_vector_type(16))) _Float16 v16h;
typedef __attribute__((ext_vector_type(8)))  float    v8f;

#define NN   8192
#define BB   8
#define LL   1024
#define EE   262144
#define HH   8
#define CC   32
#define DD   256
#define DFF  2048
#define E2   (EE + NN)   // edges + self loops

// ------------------------------ utility ------------------------------------

__global__ void fill_kernel(float* __restrict__ p, float v, int n) {
    int i = blockIdx.x * blockDim.x + threadIdx.x;
    if (i < n) p[i] = v;
}

__device__ __forceinline__ void atomicMaxF(float* addr, float val) {
    // ordered-int trick: works for mixed-sign floats
    if (val >= 0.0f) atomicMax((int*)addr, __float_as_int(val));
    else             atomicMin((unsigned int*)addr, (unsigned int)__float_as_int(val));
}

// ------------------------- generic WMMA GEMM -------------------------------
// C[M,Nout] = act(A[M,K] @ W[Nout,K]^T + bias).  K % 32 == 0, M % 64 == 0,
// Nout % 64 == 0. Block = 256 threads = 8 waves; block tile 64(M) x 64(N);
// wave (r=wave&3, c=wave>>2) computes rows r*16..+16, cols c*32..+32 via two
// WMMAs sharing one A fragment.
__global__ __launch_bounds__(256) void gemm_wmma_kernel(
    const float* __restrict__ A, const float* __restrict__ W,
    const float* __restrict__ bias, float* __restrict__ Cmat,
    int M, int K, int Nout, int relu)
{
    __shared__ _Float16 As[64][34];
    __shared__ _Float16 Ws[64][34];

    const int tid  = threadIdx.x;
    const int wave = tid >> 5;
    const int lane = tid & 31;
    const int m0   = blockIdx.y * 64;
    const int n0   = blockIdx.x * 64;
    const int mrow = (wave & 3) * 16;
    const int ncol = (wave >> 2) * 32;

    v8f acc0 = {}, acc1 = {};

    const int srow = tid >> 2;          // 0..63 (A row / W row)
    const int skc  = (tid & 3) * 8;     // 0,8,16,24

    for (int k0 = 0; k0 < K; k0 += 32) {
        {   // stage A tile (64x32) as f16
            const float* src = A + (size_t)(m0 + srow) * K + k0 + skc;
            #pragma unroll
            for (int j = 0; j < 8; ++j) As[srow][skc + j] = (_Float16)src[j];
            if (k0 + 32 < K) __builtin_prefetch(src + 32, 0, 1);
        }
        {   // stage W tile (64 output rows x 32 K) as f16
            const float* src = W + (size_t)(n0 + srow) * K + k0 + skc;
            #pragma unroll
            for (int j = 0; j < 8; ++j) Ws[srow][skc + j] = (_Float16)src[j];
        }
        __syncthreads();

        // A fragment (16x32, contiguous k = 16*(lane/16) + s, same as B)
        v16h a, b0, b1;
        const int r  = mrow + (lane & 15);
        const int kb = (lane & 16) ? 16 : 0;
        #pragma unroll
        for (int j = 0; j < 8; ++j) {
            a[j]     = As[r][kb + j];
            a[8 + j] = As[r][8 + kb + j];
        }
        // B fragments (32x16 each): B[k][n] = W[n0+ncol+n][k0+k]
        const int nr  = ncol + (lane & 15);
        const int kb2 = (lane & 16) ? 16 : 0;
        #pragma unroll
        for (int h = 0; h < 16; ++h) {
            b0[h] = Ws[nr][kb2 + h];
            b1[h] = Ws[nr + 16][kb2 + h];
        }

        acc0 = __builtin_amdgcn_wmma_f32_16x16x32_f16(false, a, false, b0,
                                                      (short)0, acc0, false, false);
        acc1 = __builtin_amdgcn_wmma_f32_16x16x32_f16(false, a, false, b1,
                                                      (short)0, acc1, false, false);
        __syncthreads();
    }

    const int col0 = n0 + ncol + (lane & 15);
    const int col1 = col0 + 16;
    const float bv0 = bias ? bias[col0] : 0.0f;
    const float bv1 = bias ? bias[col1] : 0.0f;
    #pragma unroll
    for (int i = 0; i < 8; ++i) {
        const int row = m0 + mrow + i + ((lane & 16) ? 8 : 0);
        float v0 = acc0[i] + bv0;
        float v1 = acc1[i] + bv1;
        if (relu) { v0 = v0 > 0.0f ? v0 : 0.0f; v1 = v1 > 0.0f ? v1 : 0.0f; }
        Cmat[(size_t)row * Nout + col0] = v0;
        Cmat[(size_t)row * Nout + col1] = v1;
    }
}

// ----------------------- fused flash attention -----------------------------
// One wave per (16-query tile, head, batch). 32-key step: two score WMMAs
// fill a full 16x32 P tile, then P.V runs with the complete K=32 contraction.
// Mask is all-valid (every graph has exactly L nodes).
__global__ __launch_bounds__(32) void attn_kernel(
    const float* __restrict__ qkv,   // [N, 768]: q | k | v
    float* __restrict__ O)           // [N, 256]
{
    __shared__ _Float16 ps[16][34];

    const int lane = threadIdx.x;
    const int qt = blockIdx.x, hd = blockIdx.y, bb = blockIdx.z;
    const int kb  = (lane & 16) ? 16 : 0;   // A-fragment K base
    const int kb2 = (lane & 16) ? 16 : 0;   // B-fragment K base

    // Q fragment (A layout): row = lane&15, K = head dim
    const int qrow = bb * LL + qt * 16 + (lane & 15);
    const float* qp = qkv + (size_t)qrow * 768 + hd * 32;
    v16h qf;
    #pragma unroll
    for (int j = 0; j < 8; ++j) {
        qf[j]     = (_Float16)qp[kb + j];
        qf[8 + j] = (_Float16)qp[8 + kb + j];
    }

    float rm[8], rl[8];
    v8f o0 = {}, o1 = {};
    #pragma unroll
    for (int i = 0; i < 8; ++i) { rm[i] = -1e30f; rl[i] = 0.0f; }
    const float scale = 0.17677669529663687f;  // 1/sqrt(32)

    const int prow = ((lane & 16) ? 8 : 0);

    for (int kt = 0; kt < LL / 32; ++kt) {
        const int kbase = bb * LL + kt * 32;

        // K fragments (B layout): N = key (lane&15), K = head dim
        const float* kp0 = qkv + (size_t)(kbase + (lane & 15)) * 768 + DD + hd * 32 + kb2;
        const float* kp1 = kp0 + (size_t)16 * 768;
        v16h kf0, kf1;
        #pragma unroll
        for (int h = 0; h < 16; ++h) { kf0[h] = (_Float16)kp0[h]; kf1[h] = (_Float16)kp1[h]; }

        v8f s0 = {}, s1 = {};
        s0 = __builtin_amdgcn_wmma_f32_16x16x32_f16(false, qf, false, kf0,
                                                    (short)0, s0, false, false);
        s1 = __builtin_amdgcn_wmma_f32_16x16x32_f16(false, qf, false, kf1,
                                                    (short)0, s1, false, false);

        // online softmax per row across both 16-key tiles
        #pragma unroll
        for (int i = 0; i < 8; ++i) {
            float sv0 = s0[i] * scale;
            float sv1 = s1[i] * scale;
            float mt = fmaxf(sv0, sv1);
            mt = fmaxf(mt, __shfl_xor(mt, 1, 16));
            mt = fmaxf(mt, __shfl_xor(mt, 2, 16));
            mt = fmaxf(mt, __shfl_xor(mt, 4, 16));
            mt = fmaxf(mt, __shfl_xor(mt, 8, 16));
            float mn   = fmaxf(rm[i], mt);
            float corr = __expf(rm[i] - mn);
            float p0   = __expf(sv0 - mn);
            float p1   = __expf(sv1 - mn);
            float ls = p0 + p1;
            ls += __shfl_xor(ls, 1, 16);
            ls += __shfl_xor(ls, 2, 16);
            ls += __shfl_xor(ls, 4, 16);
            ls += __shfl_xor(ls, 8, 16);
            rl[i] = rl[i] * corr + ls;
            rm[i] = mn;
            o0[i] *= corr; o1[i] *= corr;
            ps[i + prow][lane & 15]      = (_Float16)p0;
            ps[i + prow][16 + (lane & 15)] = (_Float16)p1;
        }
        __syncthreads();

        // P fragment (A layout, full K=32)
        v16h pf;
        #pragma unroll
        for (int j = 0; j < 8; ++j) {
            pf[j]     = ps[lane & 15][kb + j];
            pf[8 + j] = ps[lane & 15][8 + kb + j];
        }
        // V fragments (B layout, full 32 keys): K = key-in-32, N = dim chunk
        const float* vp = qkv + (size_t)(kbase + kb2) * 768 + 2 * DD + hd * 32 + (lane & 15);
        v16h vf0, vf1;
        #pragma unroll
        for (int h = 0; h < 16; ++h) {
            vf0[h] = (_Float16)vp[(size_t)h * 768];
            vf1[h] = (_Float16)vp[(size_t)h * 768 + 16];
        }
        o0 = __builtin_amdgcn_wmma_f32_16x16x32_f16(false, pf, false, vf0,
                                                    (short)0, o0, false, false);
        o1 = __builtin_amdgcn_wmma_f32_16x16x32_f16(false, pf, false, vf1,
                                                    (short)0, o1, false, false);
        __syncthreads();
    }

    #pragma unroll
    for (int i = 0; i < 8; ++i) {
        const float inv = 1.0f / rl[i];
        const int row = bb * LL + qt * 16 + i + prow;
        O[(size_t)row * DD + hd * 32 + (lane & 15)]      = o0[i] * inv;
        O[(size_t)row * DD + hd * 32 + 16 + (lane & 15)] = o1[i] * inv;
    }
}

// --------------------------- GAT edge softmax ------------------------------

__global__ void att_coef_kernel(const float* __restrict__ h,
                                const float* __restrict__ att_s,
                                const float* __restrict__ att_d,
                                float* __restrict__ aS, float* __restrict__ aD)
{
    int t = blockIdx.x * blockDim.x + threadIdx.x;
    if (t >= NN * HH) return;
    int n = t >> 3, hh = t & 7;
    const float* hp = h + (size_t)n * DD + hh * CC;
    const float* as = att_s + hh * CC;
    const float* ad = att_d + hh * CC;
    float s = 0.0f, d = 0.0f;
    #pragma unroll 8
    for (int c = 0; c < CC; ++c) { float v = hp[c]; s += v * as[c]; d += v * ad[c]; }
    aS[t] = s; aD[t] = d;
}

__device__ __forceinline__ void edge_sd(const int* src, const int* dst, int e, int& s, int& d) {
    if (e < EE) { s = src[e]; d = dst[e]; } else { s = e - EE; d = s; }  // self loop
}

__global__ void edge_max_kernel(const int* __restrict__ src, const int* __restrict__ dst,
                                const float* __restrict__ aS, const float* __restrict__ aD,
                                float* __restrict__ m)
{
    int t = blockIdx.x * blockDim.x + threadIdx.x;
    if (t >= E2 * HH) return;
    int e = t >> 3, hh = t & 7, s, d;
    edge_sd(src, dst, e, s, d);
    float v = aS[s * HH + hh] + aD[d * HH + hh];
    v = v > 0.0f ? v : 0.2f * v;   // leaky_relu(0.2)
    atomicMaxF(&m[d * HH + hh], v);
}

__global__ void edge_exp_kernel(const int* __restrict__ src, const int* __restrict__ dst,
                                const float* __restrict__ aS, const float* __restrict__ aD,
                                const float* __restrict__ m, float* __restrict__ ssum,
                                float* __restrict__ enum_buf)
{
    int t = blockIdx.x * blockDim.x + threadIdx.x;
    if (t >= E2 * HH) return;
    int e = t >> 3, hh = t & 7, s, d;
    edge_sd(src, dst, e, s, d);
    float v = aS[s * HH + hh] + aD[d * HH + hh];
    v = v > 0.0f ? v : 0.2f * v;
    float p = __expf(v - m[d * HH + hh]);
    enum_buf[t] = p;
    atomicAdd(&ssum[d * HH + hh], p);
}

__global__ __launch_bounds__(256) void edge_scatter_kernel(
    const int* __restrict__ src, const int* __restrict__ dst,
    const float* __restrict__ enum_buf, const float* __restrict__ ssum,
    const float* __restrict__ hfeat, float* __restrict__ out)
{
    int e = blockIdx.x;
    int c = threadIdx.x;         // channel 0..255
    int s, d;
    edge_sd(src, dst, e, s, d);
    int hh = c >> 5;
    float alpha = enum_buf[e * HH + hh] / ssum[d * HH + hh];
    atomicAdd(&out[(size_t)d * DD + c], hfeat[(size_t)s * DD + c] * alpha);
}

__global__ void bias_relu_kernel(const float* __restrict__ in,
                                 const float* __restrict__ bias,
                                 float* __restrict__ out, int n)
{
    int i = blockIdx.x * blockDim.x + threadIdx.x;
    if (i < n) { float v = in[i] + bias[i & (DD - 1)]; out[i] = v > 0.0f ? v : 0.0f; }
}

// ------------------------- residual LayerNorm ------------------------------
// y = LN(x + r) * w + b ; one wave per 256-wide row (8 rows / block)
__global__ __launch_bounds__(256) void ln_kernel(
    const float* __restrict__ x, const float* __restrict__ r,
    const float* __restrict__ w, const float* __restrict__ b,
    float* __restrict__ y)
{
    int wave = threadIdx.x >> 5, lane = threadIdx.x & 31;
    int row = blockIdx.x * 8 + wave;
    const float* xr = x + (size_t)row * DD;
    const float* rr = r + (size_t)row * DD;
    float v[8]; float sum = 0.0f;
    #pragma unroll
    for (int j = 0; j < 8; ++j) { v[j] = xr[lane + 32 * j] + rr[lane + 32 * j]; sum += v[j]; }
    #pragma unroll
    for (int off = 1; off < 32; off <<= 1) sum += __shfl_xor(sum, off, 32);
    float mu = sum * (1.0f / DD);
    float vs = 0.0f;
    #pragma unroll
    for (int j = 0; j < 8; ++j) { float dd = v[j] - mu; vs += dd * dd; }
    #pragma unroll
    for (int off = 1; off < 32; off <<= 1) vs += __shfl_xor(vs, off, 32);
    float rs = rsqrtf(vs * (1.0f / DD) + 1e-5f);
    #pragma unroll
    for (int j = 0; j < 8; ++j) {
        int c = lane + 32 * j;
        y[(size_t)row * DD + c] = (v[j] - mu) * rs * w[c] + b[c];
    }
}

// ----------------------------- prediction ----------------------------------

__global__ void pred_kernel(const float* __restrict__ h, const float* __restrict__ pw,
                            const float* __restrict__ pb, float* __restrict__ out)
{
    int t = blockIdx.x * blockDim.x + threadIdx.x;
    if (t >= NN * 4) return;
    int n = t >> 2, o = t & 3;
    if (o >= 3) return;
    const float* hp = h + (size_t)n * DD;
    const float* wp = pw + o * DD;
    float s = pb[o];
    #pragma unroll 8
    for (int c = 0; c < DD; ++c) s += hp[c] * wp[c];
    out[n * 3 + o] = s;
}

// ------------------------------- driver ------------------------------------

extern "C" void kernel_launch(void* const* d_in, const int* in_sizes, int n_in,
                              void* d_out, int out_size, void* d_ws, size_t ws_size,
                              hipStream_t stream)
{
    (void)in_sizes; (void)n_in; (void)out_size; (void)ws_size;
    const float* x        = (const float*)d_in[0];
    const int*   ei       = (const int*)  d_in[1];   // [2,E]
    const float* gat_W0   = (const float*)d_in[3];   // [D,IN]
    const float* gat_W12  = (const float*)d_in[4];   // [2,D,D]
    const float* att_src  = (const float*)d_in[5];   // [3,H,C]
    const float* att_dst  = (const float*)d_in[6];
    const float* gat_bias = (const float*)d_in[7];   // [3,D]
    const float* qkv_w    = (const float*)d_in[8];   // [2,3D,D]
    const float* qkv_b    = (const float*)d_in[9];
    const float* out_w    = (const float*)d_in[10];  // [2,D,D]
    const float* out_b    = (const float*)d_in[11];
    const float* ln1w     = (const float*)d_in[12];
    const float* ln1b     = (const float*)d_in[13];
    const float* ln2w     = (const float*)d_in[14];
    const float* ln2b     = (const float*)d_in[15];
    const float* f1w      = (const float*)d_in[16];  // [2,DFF,D]
    const float* f1b      = (const float*)d_in[17];
    const float* f2w      = (const float*)d_in[18];  // [2,D,DFF]
    const float* f2b      = (const float*)d_in[19];
    const float* pred_w   = (const float*)d_in[20];  // [3,D]
    const float* pred_b   = (const float*)d_in[21];

    const int* srcp = ei;
    const int* dstp = ei + EE;

    // workspace layout (floats)
    const size_t SZH = (size_t)NN * DD;          // 2M floats
    float* hA  = (float*)d_ws;                   // current activation
    float* hB  = hA + SZH;                       // gemm output / residual
    float* hC  = hB + SZH;                       // attn out / gat aggregation
    float* aS  = hC + SZH;                       // [N,H]
    float* aD  = aS + (size_t)NN * HH;
    float* mM  = aD + (size_t)NN * HH;
    float* sS  = mM + (size_t)NN * HH;
    float* big = sS + (size_t)NN * HH;           // 16M floats: enum / qkv / ffn-mid

    const dim3 blk256(256);
    auto grid1 = [](size_t n) { return dim3((unsigned)((n + 255) / 256)); };

    // ------------------------- 3 GAT layers -------------------------------
    for (int layer = 0; layer < 3; ++layer) {
        const float* Ain = (layer == 0) ? x : hA;
        const int    K   = (layer == 0) ? 64 : DD;
        const float* Wt  = (layer == 0) ? gat_W0 : (gat_W12 + (size_t)(layer - 1) * DD * DD);

        // h = Ain @ W^T  (no bias/act; bias added after aggregation)
        gemm_wmma_kernel<<<dim3(DD / 64, NN / 64), blk256, 0, stream>>>(
            Ain, Wt, nullptr, hB, NN, K, DD, 0);

        att_coef_kernel<<<grid1((size_t)NN * HH), blk256, 0, stream>>>(
            hB, att_src + layer * HH * CC, att_dst + layer * HH * CC, aS, aD);

        fill_kernel<<<grid1((size_t)NN * HH), blk256, 0, stream>>>(mM, -1e30f, NN * HH);
        fill_kernel<<<grid1((size_t)NN * HH), blk256, 0, stream>>>(sS, 0.0f, NN * HH);
        fill_kernel<<<grid1(SZH), blk256, 0, stream>>>(hC, 0.0f, (int)SZH);

        edge_max_kernel<<<grid1((size_t)E2 * HH), blk256, 0, stream>>>(srcp, dstp, aS, aD, mM);
        edge_exp_kernel<<<grid1((size_t)E2 * HH), blk256, 0, stream>>>(srcp, dstp, aS, aD, mM, sS, big);
        edge_scatter_kernel<<<dim3(E2), blk256, 0, stream>>>(srcp, dstp, big, sS, hB, hC);

        bias_relu_kernel<<<grid1(SZH), blk256, 0, stream>>>(hC, gat_bias + layer * DD, hA, (int)SZH);
    }

    // --------------------- 2 transformer layers ---------------------------
    for (int i = 0; i < 2; ++i) {
        // QKV: [N,768]
        gemm_wmma_kernel<<<dim3(3 * DD / 64, NN / 64), blk256, 0, stream>>>(
            hA, qkv_w + (size_t)i * 3 * DD * DD, qkv_b + (size_t)i * 3 * DD, big, NN, DD, 3 * DD, 0);

        // fused attention -> hC
        attn_kernel<<<dim3(LL / 16, HH, BB), dim3(32), 0, stream>>>(big, hC);

        // out projection
        gemm_wmma_kernel<<<dim3(DD / 64, NN / 64), blk256, 0, stream>>>(
            hC, out_w + (size_t)i * DD * DD, out_b + (size_t)i * DD, hB, NN, DD, DD, 0);

        // x = LN(x + proj)
        ln_kernel<<<dim3(NN / 8), blk256, 0, stream>>>(hA, hB, ln1w + i * DD, ln1b + i * DD, hA);

        // FFN
        gemm_wmma_kernel<<<dim3(DFF / 64, NN / 64), blk256, 0, stream>>>(
            hA, f1w + (size_t)i * DFF * DD, f1b + (size_t)i * DFF, big, NN, DD, DFF, 1);
        gemm_wmma_kernel<<<dim3(DD / 64, NN / 64), blk256, 0, stream>>>(
            big, f2w + (size_t)i * DD * DFF, f2b + (size_t)i * DD, hB, NN, DFF, DD, 0);

        ln_kernel<<<dim3(NN / 8), blk256, 0, stream>>>(hA, hB, ln2w + i * DD, ln2b + i * DD, hA);
    }

    // ----------------------------- head -----------------------------------
    pred_kernel<<<grid1((size_t)NN * 4), blk256, 0, stream>>>(hA, pred_w, pred_b, (float*)d_out);
}
